// Str2MSA_11115375362521
// MI455X (gfx1250) — matrix operand order repaired and checked by reading, not compiled, not run on hardware
//
#include <hip/hip_runtime.h>
#include <hip/hip_bf16.h>

// Str2MSA for MI455X (gfx1250): all heavy GEMMs via v_wmma_f32_16x16x32_bf16
// (fp32 accumulate). B operands stored in fragment-ready swizzled layout so
// each lane loads its 16 bf16 B elements as contiguous 32B (2x b128 loads).

typedef __bf16 bf16;
typedef __attribute__((ext_vector_type(16))) __bf16 v16bf;
typedef __attribute__((ext_vector_type(8)))  float  v8f;

#define LSEQ 512
#define NSEQ 256
#define NH   4
#define DM   64
#define DKD  32
#define DST  32
#define DFF  256
#define HK   128   // NH * DKD
#define LNEPS 1e-5f

// ---- WMMA fragment helpers (ISA 7.12.2, wave32) ----
// A: 16x32 bf16. lane m = L&15; lanes<16 hold K {0..7,16..23}, lanes>=16 hold
// K {8..15,24..31}; each VGPR packs (k,k+1). Runs of 8 contiguous bf16 ->
// compiler merges to b128 loads.
static __device__ __forceinline__ v16bf packA(const bf16* base, int ldm, int lane) {
  int m = lane & 15, hi = lane >> 4;
  const bf16* row = base + (size_t)m * ldm + hi * 8;
  v16bf a;
#pragma unroll
  for (int p = 0; p < 8; ++p) {
    int k0 = ((p & 3) << 1) + ((p & 4) << 2); // p<4: 2p ; p>=4: 16+2(p-4)
    a[2 * p]     = row[k0];
    a[2 * p + 1] = row[k0 + 1];
  }
  return a;
}

// B swizzle: a 32x16 B tile is stored as [lane][16] with each lane's elements
// contiguous. Tile stride = 512 elements (1 KB). Consumer load is one aligned
// 32-byte vector per lane.
static __device__ __forceinline__ size_t bswz_idx(int kt, int nt, int NT, int kl,
                                                  int nn) {
  int hi = (kl >> 3) & 1, s = (kl & 7) | ((kl >> 4) << 3);
  int lane = hi * 16 + nn;
  return ((((size_t)kt * NT + nt) * 32 + lane) << 4) + s;
}
static __device__ __forceinline__ v16bf loadBswz(const bf16* swz, int kt, int nt,
                                                 int NT, int lane) {
  return *(const v16bf*)(swz + (((size_t)kt * NT + nt) * 32 + lane) * 16);
}
// C/D f32: element (m = r + 8*(lane>>4), n = lane&15) in acc[r].

static __device__ __forceinline__ v8f wmma_bf16(v16bf a, v16bf b, v8f c) {
  return __builtin_amdgcn_wmma_f32_16x16x32_bf16(false, a, false, b, (short)0, c,
                                                 false, false);
}

// ---------- kernel: convert weights to bf16, fragment-swizzled ----------
static __device__ __forceinline__ void swz_store(const float* W, bf16* dst, int N,
                                                 int idx) {
  int k = idx / N, n = idx % N;
  dst[bswz_idx(k >> 5, n >> 4, N >> 4, k & 31, n & 15)] = (bf16)W[idx];
}
__global__ void k_w2bf(const float* __restrict__ Wv, const float* __restrict__ Wo,
                       const float* __restrict__ W1, const float* __restrict__ W2,
                       bf16* Wv_b, bf16* Wo_b, bf16* W1_b, bf16* W2_b) {
  int idx = blockIdx.x * 256 + threadIdx.x;
  if (idx < 4096)       swz_store(Wv, Wv_b, DM, idx);
  else if (idx < 8192)  swz_store(Wo, Wo_b, DM, idx - 4096);
  else if (idx < 24576) swz_store(W1, W1_b, DFF, idx - 8192);
  else if (idx < 40960) swz_store(W2, W2_b, DM, idx - 24576);
}

// ---------- kernel: st = LN(state); q = st@Wq*scale; k = st@Wk ----------
__global__ void k_qk(const float* __restrict__ state, const float* __restrict__ ns_g,
                     const float* __restrict__ ns_b,
                     const float* __restrict__ Wq, const float* __restrict__ bq,
                     const float* __restrict__ Wk, const float* __restrict__ bk,
                     float* __restrict__ q, float* __restrict__ k) {
  __shared__ float stn[DST];
  __shared__ float mv[2];
  int i = blockIdx.x, t = threadIdx.x;
  const float* xr = state + (size_t)i * DST;
  if (t == 0) {
    float m = 0.f;
    for (int d = 0; d < DST; ++d) m += xr[d];
    m *= (1.f / DST);
    float v = 0.f;
    for (int d = 0; d < DST; ++d) { float z = xr[d] - m; v += z * z; }
    v *= (1.f / DST);
    mv[0] = m; mv[1] = rsqrtf(v + LNEPS);
  }
  __syncthreads();
  if (t < DST) stn[t] = (xr[t] - mv[0]) * mv[1] * ns_g[t] + ns_b[t];
  __syncthreads();
  float qv = bq[t], kv = bk[t];
#pragma unroll
  for (int d = 0; d < DST; ++d) {
    float s = stn[d];
    qv += s * Wq[d * HK + t];
    kv += s * Wk[d * HK + t];
  }
  const float scale = 0.17677669529663687f; // 1/sqrt(D_K)
  q[(size_t)i * HK + t] = qv * scale;
  k[(size_t)i * HK + t] = kv;
}

// ---------- kernel: scores + distance mask + softmax -> attn bf16 [h][i][j] ----------
__global__ void k_attn(const float* __restrict__ q, const float* __restrict__ k,
                       const float* __restrict__ Ca, bf16* __restrict__ attn) {
  __shared__ float qs[DKD];
  __shared__ float sc[LSEQ];
  __shared__ float red[256];
  int i = blockIdx.x, h = blockIdx.y, t = threadIdx.x;
  if (t < DKD) qs[t] = q[(size_t)i * HK + h * DKD + t];
  __syncthreads();
  const float bins[NH] = {0.4f, 0.8f, 1.2f, 1.6f}; // DISTBIN / TRANS
  float binh = bins[h];
  float cx = Ca[i * 3], cy = Ca[i * 3 + 1], cz = Ca[i * 3 + 2];
  float lmax = -3.4028235e38f;
  for (int j = t; j < LSEQ; j += 256) {
    const float* kr = k + (size_t)j * HK + h * DKD;
    float s = 0.f;
#pragma unroll
    for (int d = 0; d < DKD; ++d) s += qs[d] * kr[d];
    float dx = cx - Ca[j * 3], dy = cy - Ca[j * 3 + 1], dz = cz - Ca[j * 3 + 2];
    float dist = sqrtf(fmaxf(dx * dx + dy * dy + dz * dz, 1e-12f));
    // mask_s < 0.5  <=>  dist > bin  -> finfo.min
    if (dist > binh) s = -3.4028235e38f;
    sc[j] = s;
    lmax = fmaxf(lmax, s);
  }
  red[t] = lmax; __syncthreads();
  for (int w = 128; w > 0; w >>= 1) {
    if (t < w) red[t] = fmaxf(red[t], red[t + w]);
    __syncthreads();
  }
  float m = red[0]; __syncthreads();
  float lsum = 0.f;
  for (int j = t; j < LSEQ; j += 256) {
    float p = __expf(sc[j] - m);  // masked -> exp(-huge) = 0
    sc[j] = p; lsum += p;
  }
  red[t] = lsum; __syncthreads();
  for (int w = 128; w > 0; w >>= 1) {
    if (t < w) red[t] += red[t + w];
    __syncthreads();
  }
  float inv = 1.f / red[0];
  bf16* arow = attn + ((size_t)h * LSEQ + i) * LSEQ;
  for (int j = t; j < LSEQ; j += 256) arow[j] = (bf16)(sc[j] * inv);
}

// ---------- kernel: v = LN(msa,n1)@Wv + bv -> B-swizzled bf16 ----------
// Output layout: vswz[n][kt][h][lane][16] (kt = j/32, h = c/16), so k_attnv's
// B fragments are contiguous per lane.
__global__ void k_vproj(const float* __restrict__ msa, const float* __restrict__ n1_g,
                        const float* __restrict__ n1_b, const bf16* __restrict__ Wv_b,
                        const float* __restrict__ bv, bf16* __restrict__ vswz) {
  __shared__ bf16 ta[16][DM];
  int j0 = blockIdx.x * 16, n = blockIdx.y;
  int t = threadIdx.x, lane = t & 31, wv = t >> 5;
  if (t < 16) {
    const float* xr = msa + ((size_t)n * LSEQ + j0 + t) * DM;
    float m = 0.f;
    for (int c = 0; c < DM; ++c) m += xr[c];
    m *= (1.f / DM);
    float v = 0.f;
    for (int c = 0; c < DM; ++c) { float z = xr[c] - m; v += z * z; }
    v *= (1.f / DM);
    float r = rsqrtf(v + LNEPS);
    for (int c = 0; c < DM; ++c)
      ta[t][c] = (bf16)((xr[c] - m) * r * n1_g[c] + n1_b[c]);
  }
  __syncthreads();
  v8f acc = {};
#pragma unroll
  for (int kk = 0; kk < DM; kk += 32) {
    v16bf a = packA(&ta[0][0] + kk, DM, lane);
    v16bf b = loadBswz(Wv_b, kk >> 5, wv, NH, lane);
    acc = wmma_bf16(a, b, acc);
  }
  int nn = lane & 15, hi = lane >> 4;
  int h = wv;                     // output col tile = head
  float bvv = bv[h * 16 + nn];
  bf16* nbase = vswz + (size_t)n * LSEQ * DM;  // 16 kt * 4 h * 512
#pragma unroll
  for (int r = 0; r < 8; ++r) {
    int j = j0 + r + hi * 8;      // global row
    int kt = j >> 5, kl = j & 31;
    nbase[bswz_idx(kt, h, NH, kl, nn)] = (bf16)(acc[r] + bvv);
  }
}

// ---------- kernel: out = attn@v ; out@Wo + bo + msa ; LN(.,n2) -> xn ----------
// Each wave owns one head; processes TWO n values per A-fragment load.
__global__ void k_attnv(const bf16* __restrict__ attn, const bf16* __restrict__ vswz,
                        const bf16* __restrict__ Wo_b, const float* __restrict__ bo,
                        const float* __restrict__ msa, const float* __restrict__ n2_g,
                        const float* __restrict__ n2_b, float* __restrict__ msa1,
                        bf16* __restrict__ xnbf) {
  __shared__ bf16 latt[2][16][DM];
  __shared__ float lms[2][16][DM];
  int i0 = blockIdx.x * 16, n0 = blockIdx.y * 2;
  int t = threadIdx.x, lane = t & 31, wv = t >> 5;
  int h = wv; // one head per wave
  v8f acc0 = {}, acc1 = {};
  const bf16* Abase = attn + ((size_t)h * LSEQ + i0) * LSEQ;
  const bf16* Vb0 = vswz + (size_t)n0 * LSEQ * DM;
  const bf16* Vb1 = Vb0 + (size_t)LSEQ * DM;
  for (int kt = 0; kt < LSEQ / 32; ++kt) {
    __builtin_prefetch(Abase + (kt + 1) * 32, 0, 0);
    v16bf a  = packA(Abase + kt * 32, LSEQ, lane);
    v16bf b0 = loadBswz(Vb0, kt, h, NH, lane);
    v16bf b1 = loadBswz(Vb1, kt, h, NH, lane);
    acc0 = wmma_bf16(a, b0, acc0);
    acc1 = wmma_bf16(a, b1, acc1);
  }
  int nn = lane & 15, hi = lane >> 4;
#pragma unroll
  for (int r = 0; r < 8; ++r) {
    latt[0][r + hi * 8][h * 16 + nn] = (bf16)acc0[r];
    latt[1][r + hi * 8][h * 16 + nn] = (bf16)acc1[r];
  }
  __syncthreads();
  int c0 = wv * 16;
#pragma unroll
  for (int s = 0; s < 2; ++s) {
    v8f oacc = {};
#pragma unroll
    for (int kk = 0; kk < DM; kk += 32) {
      v16bf a = packA(&latt[s][0][0] + kk, DM, lane);
      v16bf b = loadBswz(Wo_b, kk >> 5, wv, NH, lane);
      oacc = wmma_bf16(a, b, oacc);
    }
    float bov = bo[c0 + nn];
#pragma unroll
    for (int r = 0; r < 8; ++r) {
      int m = r + hi * 8;
      lms[s][m][c0 + nn] =
          oacc[r] + bov + msa[((size_t)(n0 + s) * LSEQ + i0 + m) * DM + c0 + nn];
    }
  }
  __syncthreads();
  if (t < 32) {
    int s = t >> 4, row = t & 15;
    float* xr = lms[s][row];
    float mm = 0.f;
    for (int c = 0; c < DM; ++c) mm += xr[c];
    mm *= (1.f / DM);
    float vv = 0.f;
    for (int c = 0; c < DM; ++c) { float z = xr[c] - mm; vv += z * z; }
    vv *= (1.f / DM);
    float rr = rsqrtf(vv + LNEPS);
    size_t rowo = ((size_t)(n0 + s) * LSEQ + i0 + row) * DM;
    for (int c = 0; c < DM; ++c) {
      float val = xr[c];
      msa1[rowo + c] = val;
      xnbf[rowo + c] = (bf16)((val - mm) * rr * n2_g[c] + n2_b[c]);
    }
  }
}

// ---------- kernel: fused FF + final LN ----------
__global__ void k_ff(const bf16* __restrict__ xnbf, const float* __restrict__ msa1,
                     const bf16* __restrict__ W1_b, const float* __restrict__ b1,
                     const bf16* __restrict__ W2_b, const float* __restrict__ b2,
                     const float* __restrict__ n3_g, const float* __restrict__ n3_b,
                     float* __restrict__ out) {
  __shared__ bf16 lA[16][DM];
  __shared__ bf16 lH[16][DFF];
  __shared__ float lms[16][DM];
  int i0 = blockIdx.x * 16, n = blockIdx.y;
  int t = threadIdx.x, lane = t & 31, wv = t >> 5;
  {
    const bf16* src = xnbf + ((size_t)n * LSEQ + i0) * DM;
    for (int e = t; e < 16 * DM; e += 128) (&lA[0][0])[e] = src[e];
  }
  __syncthreads();
  int nn = lane & 15, hi = lane >> 4;
  // FF1: relu(xn @ W1 + b1), each wave owns 64 hidden cols
#pragma unroll
  for (int ct = 0; ct < 4; ++ct) {
    int nt = wv * 4 + ct;         // hidden col tile (of 16)
    v8f acc = {};
#pragma unroll
    for (int kk = 0; kk < DM; kk += 32) {
      v16bf a = packA(&lA[0][0] + kk, DM, lane);
      v16bf b = loadBswz(W1_b, kk >> 5, nt, DFF / 16, lane);
      acc = wmma_bf16(a, b, acc);
    }
    float b1v = b1[nt * 16 + nn];
#pragma unroll
    for (int r = 0; r < 8; ++r)
      lH[r + hi * 8][nt * 16 + nn] = (bf16)fmaxf(acc[r] + b1v, 0.f);
  }
  __syncthreads();
  // FF2: hidden @ W2 + b2 + residual
  int c0 = wv * 16;
  v8f acc = {};
  for (int kk = 0; kk < DFF; kk += 32) {
    v16bf a = packA(&lH[0][0] + kk, DFF, lane);
    v16bf b = loadBswz(W2_b, kk >> 5, wv, NH, lane);
    acc = wmma_bf16(a, b, acc);
  }
  float b2v = b2[c0 + nn];
#pragma unroll
  for (int r = 0; r < 8; ++r) {
    int m = r + hi * 8;
    lms[m][c0 + nn] =
        acc[r] + b2v + msa1[((size_t)n * LSEQ + i0 + m) * DM + c0 + nn];
  }
  __syncthreads();
  if (t < 16) {
    float* xr = lms[t];
    float mm = 0.f;
    for (int c = 0; c < DM; ++c) mm += xr[c];
    mm *= (1.f / DM);
    float vv = 0.f;
    for (int c = 0; c < DM; ++c) { float z = xr[c] - mm; vv += z * z; }
    vv *= (1.f / DM);
    float rr = rsqrtf(vv + LNEPS);
    size_t rowo = ((size_t)n * LSEQ + i0 + t) * DM;
    for (int c = 0; c < DM; ++c)
      out[rowo + c] = (xr[c] - mm) * rr * n3_g[c] + n3_b[c];
  }
}

extern "C" void kernel_launch(void* const* d_in, const int* in_sizes, int n_in,
                              void* d_out, int out_size, void* d_ws, size_t ws_size,
                              hipStream_t stream) {
  const float* msa   = (const float*)d_in[0];
  const float* Ca    = (const float*)d_in[1];
  const float* state = (const float*)d_in[2];
  const float* ns_g  = (const float*)d_in[3];
  const float* ns_b  = (const float*)d_in[4];
  const float* n1_g  = (const float*)d_in[5];
  const float* n1_b  = (const float*)d_in[6];
  const float* Wq    = (const float*)d_in[7];
  const float* bq    = (const float*)d_in[8];
  const float* Wk    = (const float*)d_in[9];
  const float* bk    = (const float*)d_in[10];
  const float* Wv    = (const float*)d_in[11];
  const float* bv    = (const float*)d_in[12];
  const float* Wo    = (const float*)d_in[13];
  const float* bo    = (const float*)d_in[14];
  const float* n2_g  = (const float*)d_in[15];
  const float* n2_b  = (const float*)d_in[16];
  const float* W1    = (const float*)d_in[17];
  const float* b1    = (const float*)d_in[18];
  const float* W2    = (const float*)d_in[19];
  const float* b2    = (const float*)d_in[20];
  const float* n3_g  = (const float*)d_in[21];
  const float* n3_b  = (const float*)d_in[22];
  float* out = (float*)d_out;

  char* w = (char*)d_ws;
  float* q    = (float*)w; w += (size_t)LSEQ * HK * 4;
  float* kbuf = (float*)w; w += (size_t)LSEQ * HK * 4;
  bf16* attn  = (bf16*)w;  w += (size_t)NH * LSEQ * LSEQ * 2;
  bf16* vswz  = (bf16*)w;  w += (size_t)NSEQ * LSEQ * DM * 2;
  bf16* xnbf  = (bf16*)w;  w += (size_t)NSEQ * LSEQ * DM * 2;
  float* msa1 = (float*)w; w += (size_t)NSEQ * LSEQ * DM * 4;
  bf16* Wv_b  = (bf16*)w;  w += 4096 * 2;
  bf16* Wo_b  = (bf16*)w;  w += 4096 * 2;
  bf16* W1_b  = (bf16*)w;  w += 16384 * 2;
  bf16* W2_b  = (bf16*)w;  w += 16384 * 2;

  k_w2bf<<<160, 256, 0, stream>>>(Wv, Wo, W1, W2, Wv_b, Wo_b, W1_b, W2_b);
  k_qk<<<LSEQ, 128, 0, stream>>>(state, ns_g, ns_b, Wq, bq, Wk, bk, q, kbuf);
  k_attn<<<dim3(LSEQ, NH), 256, 0, stream>>>(q, kbuf, Ca, attn);
  k_vproj<<<dim3(LSEQ / 16, NSEQ), 128, 0, stream>>>(msa, n1_g, n1_b, Wv_b, bv, vswz);
  k_attnv<<<dim3(LSEQ / 16, NSEQ / 2), 128, 0, stream>>>(attn, vswz, Wo_b, bo, msa,
                                                         n2_g, n2_b, msa1, xnbf);
  k_ff<<<dim3(LSEQ / 16, NSEQ), 128, 0, stream>>>(xnbf, msa1, W1_b, b1, W2_b, b2,
                                                  n3_g, n3_b, out);
}
